// GAT_59133109732231
// MI455X (gfx1250) — matrix operand order repaired and checked by reading, not compiled, not run on hardware
//
#include <hip/hip_runtime.h>
#include <hip/hip_bf16.h>
#include <math.h>

// ---------------------------------------------------------------------------
// GAT (2x GATConv + MLP head) for MI455X / gfx1250 (wave32, WMMA).
//  - GEMMs: v_wmma_f32_16x16x32_f16 (f16 inputs, f32 accumulate)
//  - Transformed features h kept in f16 (halves edge-gather traffic; h fits L2)
//  - Edge softmax: float atomic max / add; aggregation: f32 atomic scatter-add
// ---------------------------------------------------------------------------

typedef __attribute__((ext_vector_type(16))) _Float16 v16h;
typedef __attribute__((ext_vector_type(8)))  float    v8f;

#define IN_DIM    128
#define HID       128
#define HEADS     2
#define FEATS     (HEADS * HID)   // 256
#define OUT_DIM   2
#define NEG_SLOPE 0.2f

// ---------------------------------------------------------------------------
// generic fill
// ---------------------------------------------------------------------------
__global__ void fill_kernel(float* __restrict__ p, float v, long long n) {
  long long i = (long long)blockIdx.x * blockDim.x + threadIdx.x;
  if (i < n) p[i] = v;
}

// f32 -> f16 elementwise (weights, done once per launch; tiny)
__global__ void cvt_f32_f16(const float* __restrict__ src,
                            _Float16* __restrict__ dst, long long n) {
  long long i = (long long)blockIdx.x * blockDim.x + threadIdx.x;
  if (i < n) dst[i] = (_Float16)src[i];
}

// ---------------------------------------------------------------------------
// C[M,Nc] = A[M,K] @ B[K,Nc] (+ optional bias[Nc])
// A: f32 (converted to f16 on load); B: f16; accumulate f32 via WMMA.
// storeHalf: 1 -> C is _Float16*, 0 -> C is float*.
// One wave per 16x16 output tile. Requires M%16==0, Nc%16==0, K%32==0.
//
// A fragment (16-bit A 16x32, same K order as B): lane 0-15 -> M=lane; halves 0..15
// hold K=laneHi*16..+15, contiguous (laneHi=lane>>4).
// B fragment (16-bit B KxN): lane -> N; half i holds K = laneHi*16 + i.
// C/D: VGPR r, lane L: M = r + (L>=16 ? 8 : 0), N = L&15.
// ---------------------------------------------------------------------------
__global__ void gemm_wmma(const float* __restrict__ A,
                          const _Float16* __restrict__ B,
                          const float* __restrict__ bias, void* __restrict__ Cv,
                          int M, int Nc, int K, int useBias, int storeHalf) {
  int wave   = blockIdx.x * (blockDim.x >> 5) + (threadIdx.x >> 5);
  int tilesN = Nc >> 4;
  int tiles  = (M >> 4) * tilesN;
  if (wave >= tiles) return;
  int tm = wave / tilesN;
  int tn = wave % tilesN;

  int lane   = threadIdx.x & 31;
  int laneIn = lane & 15;
  int laneHi = lane >> 4;

  int row = tm * 16 + laneIn;          // A row this lane supplies
  int col = tn * 16 + laneIn;          // B/C column this lane supplies
  const float* arow = A + (long long)row * K;

  v8f acc = {};
  for (int kk = 0; kk < K; kk += 32) {
    v16h a, b;
    const float* a0 = arow + kk + laneHi * 16;
#pragma unroll
    for (int i = 0; i < 8; ++i) {
      a[i]     = (_Float16)a0[i];        // K = kk + laneHi*16 + i
      a[i + 8] = (_Float16)a0[8 + i];    // K = kk + laneHi*16 + 8 + i
    }
    const _Float16* b0 = B + (long long)(kk + laneHi * 16) * Nc + col;
#pragma unroll
    for (int i = 0; i < 16; ++i)
      b[i] = b0[(long long)i * Nc];      // K = kk + laneHi*16 + i

    acc = __builtin_amdgcn_wmma_f32_16x16x32_f16(
        /*neg_a=*/false, a, /*neg_b=*/false, b,
        /*c_mod=*/(short)0, acc, /*reuse_a=*/false, /*reuse_b=*/false);
  }

  float bv = useBias ? bias[col] : 0.0f;
  if (storeHalf) {
    _Float16* C = (_Float16*)Cv;
#pragma unroll
    for (int r = 0; r < 8; ++r) {
      int orow = tm * 16 + r + laneHi * 8;
      C[(long long)orow * Nc + col] = (_Float16)(acc[r] + bv);
    }
  } else {
    float* C = (float*)Cv;
#pragma unroll
    for (int r = 0; r < 8; ++r) {
      int orow = tm * 16 + r + laneHi * 8;
      C[(long long)orow * Nc + col] = acc[r] + bv;
    }
  }
}

// ---------------------------------------------------------------------------
// Per-(node, head) attention logits from f16 h: asrc = <h[n,h,:], a_src[h,:]>.
// ---------------------------------------------------------------------------
__global__ void alphas_kernel(const _Float16* __restrict__ h,
                              const float* __restrict__ a_src,
                              const float* __restrict__ a_dst,
                              float* __restrict__ asrc, float* __restrict__ adst,
                              int n) {
  int i = blockIdx.x * blockDim.x + threadIdx.x;   // node*HEADS + head
  if (i >= n * HEADS) return;
  int node = i >> 1, hd = i & 1;
  const _Float16* hp = h + (long long)node * FEATS + hd * HID;
  const float* as = a_src + hd * HID;
  const float* ad = a_dst + hd * HID;
  float s0 = 0.f, s1 = 0.f;
#pragma unroll 4
  for (int c = 0; c < HID; ++c) {
    float v = (float)hp[c];
    s0 += v * as[c];
    s1 += v * ad[c];
  }
  asrc[i] = s0;
  adst[i] = s1;
}

// float atomic max via signed-max / unsigned-min ordering trick
__device__ inline void atomicMaxF(float* addr, float val) {
  int iv = __float_as_int(val);
  if (iv >= 0) atomicMax((int*)addr, iv);
  else         atomicMin((unsigned int*)addr, (unsigned int)iv);
}

__device__ inline void edge_sd(const long long* __restrict__ ei, long long e,
                               long long E, int& s, int& d) {
  if (e < E) { s = (int)ei[e]; d = (int)ei[E + e]; }
  else       { s = d = (int)(e - E); }              // self-loop
}

// ---------------------------------------------------------------------------
// Pass 1: per-edge leaky-relu logit, store it, segment-max into m[dst,head].
// ---------------------------------------------------------------------------
__global__ void edge_logit_max(const long long* __restrict__ ei,
                               const float* __restrict__ asrc,
                               const float* __restrict__ adst,
                               float* __restrict__ elog, float* __restrict__ m,
                               long long E, int n) {
  long long i = (long long)blockIdx.x * blockDim.x + threadIdx.x; // edge*2+head
  long long Etot = E + n;
  if (i >= Etot * HEADS) return;
  long long e = i >> 1;
  int hd = (int)(i & 1);
  int s, d;
  edge_sd(ei, e, E, s, d);
  float v = asrc[(long long)s * HEADS + hd] + adst[(long long)d * HEADS + hd];
  v = (v > 0.f) ? v : NEG_SLOPE * v;
  elog[i] = v;
  atomicMaxF(&m[(long long)d * HEADS + hd], v);
}

// ---------------------------------------------------------------------------
// Pass 2: elog <- exp(logit - m[dst]); segment-sum into denom[dst,head].
// ---------------------------------------------------------------------------
__global__ void edge_exp_sum(const long long* __restrict__ ei,
                             float* __restrict__ elog,
                             const float* __restrict__ m,
                             float* __restrict__ denom,
                             long long E, int n) {
  long long i = (long long)blockIdx.x * blockDim.x + threadIdx.x;
  long long Etot = E + n;
  if (i >= Etot * HEADS) return;
  long long e = i >> 1;
  int hd = (int)(i & 1);
  int s, d;
  edge_sd(ei, e, E, s, d);
  float ex = __expf(elog[i] - m[(long long)d * HEADS + hd]);
  elog[i] = ex;
  atomicAdd(&denom[(long long)d * HEADS + hd], ex);
}

// ---------------------------------------------------------------------------
// Pass 3: weighted scatter-add. One block (256 thr) per edge; thread c handles
// feature c (head = c>>7). out[dst,c] += h16[src,c] * alpha(e,head).
// ---------------------------------------------------------------------------
__global__ void edge_aggregate(const long long* __restrict__ ei,
                               const _Float16* __restrict__ hlin,
                               const float* __restrict__ elog,
                               const float* __restrict__ denom,
                               float* __restrict__ out,
                               long long E, int n) {
  long long e = blockIdx.x;
  __shared__ int sd[2];
  if (threadIdx.x == 0) {
    int s, d;
    edge_sd(ei, e, E, s, d);
    sd[0] = s; sd[1] = d;
  }
  __syncthreads();
  int s = sd[0], d = sd[1];
  int c  = threadIdx.x;
  int hd = c >> 7;
  float w = elog[e * HEADS + hd] /
            (denom[(long long)d * HEADS + hd] + 1e-16f);
  atomicAdd(&out[(long long)d * FEATS + c],
            (float)hlin[(long long)s * FEATS + c] * w);
}

// ---------------------------------------------------------------------------
// out[i] = relu(out[i] + bias[i % FEATS])
// ---------------------------------------------------------------------------
__global__ void bias_relu(float* __restrict__ h, const float* __restrict__ b,
                          long long total) {
  long long i = (long long)blockIdx.x * blockDim.x + threadIdx.x;
  if (i >= total) return;
  int col = (int)(i & (FEATS - 1));
  h[i] = fmaxf(h[i] + b[col], 0.f);
}

// ---------------------------------------------------------------------------
// Final tiny linear (128 -> 2) + sigmoid. One thread per node.
// ---------------------------------------------------------------------------
__global__ void final_kernel(const float* __restrict__ hp,
                             const float* __restrict__ Wp2,
                             const float* __restrict__ bp2,
                             float* __restrict__ out, int n) {
  int i = blockIdx.x * blockDim.x + threadIdx.x;
  if (i >= n) return;
  const float* hr = hp + (long long)i * HID;
  float s0 = bp2[0], s1 = bp2[1];
#pragma unroll 4
  for (int j = 0; j < HID; ++j) {
    float v = hr[j];
    s0 += v * Wp2[j * OUT_DIM + 0];
    s1 += v * Wp2[j * OUT_DIM + 1];
  }
  out[(long long)i * OUT_DIM + 0] = 1.f / (1.f + __expf(-s0));
  out[(long long)i * OUT_DIM + 1] = 1.f / (1.f + __expf(-s1));
}

// ---------------------------------------------------------------------------
// launch
// ---------------------------------------------------------------------------
extern "C" void kernel_launch(void* const* d_in, const int* in_sizes, int n_in,
                              void* d_out, int out_size, void* d_ws, size_t ws_size,
                              hipStream_t stream) {
  const float*     x      = (const float*)d_in[0];
  const long long* ei     = (const long long*)d_in[1];   // int64 (2,E) row-major
  const float*     W1     = (const float*)d_in[2];
  const float*     a_src1 = (const float*)d_in[3];
  const float*     a_dst1 = (const float*)d_in[4];
  const float*     b1     = (const float*)d_in[5];
  const float*     W2     = (const float*)d_in[6];
  const float*     a_src2 = (const float*)d_in[7];
  const float*     a_dst2 = (const float*)d_in[8];
  const float*     b2     = (const float*)d_in[9];
  const float*     Wp1    = (const float*)d_in[10];
  const float*     bp1    = (const float*)d_in[11];
  const float*     Wp2    = (const float*)d_in[12];
  const float*     bp2    = (const float*)d_in[13];

  const int       n    = in_sizes[0] / IN_DIM;     // 100000
  const long long E    = in_sizes[1] / 2;          // 800000
  const long long Etot = E + n;                    // with self-loops

  // ---- workspace layout ----------------------------------------------------
  // hlinH region: n*FEATS f16 (== n*HID f32 bytes) -> later reused as hp1 f32
  char* wsb = (char*)d_ws;
  _Float16* hlinH = (_Float16*)wsb;                         // n*FEATS*2 B
  float*    hp1   = (float*)wsb;                            // reuse (n*HID*4 B)
  char* p = wsb + (size_t)n * FEATS * sizeof(_Float16);
  float* outB  = (float*)p;  p += (size_t)n * FEATS * sizeof(float);
  _Float16* W1h  = (_Float16*)p; p += (size_t)IN_DIM * FEATS * sizeof(_Float16);
  _Float16* W2h  = (_Float16*)p; p += (size_t)FEATS  * FEATS * sizeof(_Float16);
  _Float16* Wp1h = (_Float16*)p; p += (size_t)FEATS  * HID   * sizeof(_Float16);
  float* asrc  = (float*)p; p += (size_t)n * HEADS * sizeof(float);
  float* adst  = (float*)p; p += (size_t)n * HEADS * sizeof(float);
  float* mbuf  = (float*)p; p += (size_t)n * HEADS * sizeof(float);
  float* denom = (float*)p; p += (size_t)n * HEADS * sizeof(float);
  float* elog  = (float*)p;  // Etot*HEADS floats

  const int T = 256;
  auto blks = [](long long work, int t) { return (unsigned)((work + t - 1) / t); };

  // weights -> f16 (once per launch; tiny)
  cvt_f32_f16<<<blks(IN_DIM * FEATS, T), T, 0, stream>>>(W1, W1h, IN_DIM * FEATS);
  cvt_f32_f16<<<blks(FEATS  * FEATS, T), T, 0, stream>>>(W2, W2h, FEATS * FEATS);
  cvt_f32_f16<<<blks(FEATS  * HID,   T), T, 0, stream>>>(Wp1, Wp1h, FEATS * HID);

  auto gat_layer = [&](const float* in, int K, const _Float16* Wh,
                       const float* avs, const float* avd, const float* bvec) {
    // h = in @ W   (f16 output; reference adds bias after aggregation)
    int tiles = (n / 16) * (FEATS / 16);
    gemm_wmma<<<blks(tiles, 8), T, 0, stream>>>(in, Wh, nullptr, hlinH,
                                                n, FEATS, K, 0, 1);
    // per-node logits
    alphas_kernel<<<blks((long long)n * HEADS, T), T, 0, stream>>>(
        hlinH, avs, avd, asrc, adst, n);
    // init segment buffers + output accumulator
    fill_kernel<<<blks((long long)n * HEADS, T), T, 0, stream>>>(
        mbuf, -INFINITY, (long long)n * HEADS);
    fill_kernel<<<blks((long long)n * HEADS, T), T, 0, stream>>>(
        denom, 0.f, (long long)n * HEADS);
    fill_kernel<<<blks((long long)n * FEATS, T), T, 0, stream>>>(
        outB, 0.f, (long long)n * FEATS);
    // softmax over incoming edges
    edge_logit_max<<<blks(Etot * HEADS, T), T, 0, stream>>>(ei, asrc, adst,
                                                            elog, mbuf, E, n);
    edge_exp_sum  <<<blks(Etot * HEADS, T), T, 0, stream>>>(ei, elog, mbuf,
                                                            denom, E, n);
    // weighted scatter-add (gathers f16 h, accumulates f32)
    edge_aggregate<<<(unsigned)Etot, T, 0, stream>>>(ei, hlinH, elog, denom,
                                                     outB, E, n);
    // + bias, ReLU
    bias_relu<<<blks((long long)n * FEATS, T), T, 0, stream>>>(
        outB, bvec, (long long)n * FEATS);
  };

  // layer 1: x (n x 128) -> h1 in outB
  gat_layer(x, IN_DIM, W1h, a_src1, a_dst1, b1);
  // layer 2: h1 (outB) -> hlinH, aggregate back into outB (h1 fully consumed
  // by the GEMM before outB is zeroed; stream order makes the reuse safe)
  gat_layer(outB, FEATS, W2h, a_src2, a_dst2, b2);

  // post_mp: hp1 = h2 @ Wp1 + bp1 (f32 output, reuses hlinH bytes)
  int tiles_p = (n / 16) * (HID / 16);
  gemm_wmma<<<blks(tiles_p, 8), T, 0, stream>>>(outB, Wp1h, bp1, hp1,
                                                n, HID, FEATS, 1, 0);
  // final 128->2 + sigmoid
  final_kernel<<<blks(n, T), T, 0, stream>>>(hp1, Wp2, bp2, (float*)d_out, n);
}
